// LSTMNet_712964571502
// MI455X (gfx1250) — matrix operand order repaired and checked by reading, not compiled, not run on hardware
//
#include <hip/hip_runtime.h>

typedef __attribute__((ext_vector_type(16))) _Float16    v16h;
typedef __attribute__((ext_vector_type(8)))  float       v8f;
typedef __attribute__((ext_vector_type(4)))  unsigned int v4u;

#define SEQ    600
#define BATCH  4096
#define INDIM  6
#define HDIM   30
#define OUTDIM 61
#define G4H    120          // 4*H
#define ZS     72           // z row stride in halves (144 B, keeps 16B alignment, staggers banks)
#define GS     132          // gate-buffer row stride in floats

union AB16 { v16h v; v4u u[2]; };

__device__ __forceinline__ float fast_sigmoid(float x) {
    return __builtin_amdgcn_rcpf(1.0f + __expf(-x));
}
__device__ __forceinline__ float fast_tanh(float x) {
    return 1.0f - 2.0f * __builtin_amdgcn_rcpf(1.0f + __expf(2.0f * x));
}

__global__ __launch_bounds__(256)
void lstm_fused_kernel(const float* __restrict__ X,
                       const float* __restrict__ W_ih,
                       const float* __restrict__ W_hh,
                       const float* __restrict__ b_ih,
                       const float* __restrict__ b_hh,
                       const float* __restrict__ W1,
                       const float* __restrict__ b1,
                       const float* __restrict__ W2,
                       const float* __restrict__ b2,
                       float* __restrict__ out)
{
    // z: [16 rows][K=64 padded]: cols 0..29 = h (f16), 30..35 = x_t (f16), rest 0
    __shared__ __align__(16) _Float16 zsh[16 * ZS];
    // g: [16 rows][128 gate cols] f32; also reused for FC epilogue
    __shared__ __align__(16) float    gsh[16 * GS];

    const int tid  = threadIdx.x;
    const int lane = tid & 31;
    const int wv   = tid >> 5;                 // wave 0..7 -> N tile
    const int b0   = blockIdx.x * 16;          // batch tile base

    // ------------------------------------------------------------------
    // One-time: build persistent B (weight) fragments in VGPRs.
    // B is [K=64][N=128]: rows 0..29 = W_hh^T, rows 30..35 = W_ih^T, else 0.
    // 16-bit B layout: lane = N (mod 16); lanes<16 hold K=0..7,16..23, lanes>=16 K=8..15,24..31
    // (frag f adds f*32 to K).
    // ------------------------------------------------------------------
    const int n  = wv * 16 + (lane & 15);      // gate column 0..127
    const int kb = (lane < 16) ? 0 : 16;
    AB16 bfrag[2];
#pragma unroll
    for (int f = 0; f < 2; ++f) {
#pragma unroll
        for (int e = 0; e < 16; ++e) {
            const int k = f * 32 + ((e < 8) ? (kb >> 1) + e : 16 + (kb >> 1) + (e - 8));
            float w = 0.0f;
            if (n < G4H) {
                if (k < HDIM)                   w = W_hh[n * HDIM + k];
                else if (k < HDIM + INDIM)      w = W_ih[n * INDIM + (k - HDIM)];
            }
            bfrag[f].v[e] = (_Float16)w;
        }
    }
    const float bias_n = (n < G4H) ? (b_ih[n] + b_hh[n]) : 0.0f;

    // zero the z buffer (pad region stays zero forever)
    for (int i = tid; i < 16 * ZS; i += 256) zsh[i] = (_Float16)0.0f;

    // gate-element ownership: element e = m*30 + j, 480 total
    const int e1 = tid + 256;
    float c0 = 0.0f, c1 = 0.0f, h0f = 0.0f, h1f = 0.0f;

    // load x_0
    float xreg = 0.0f;
    if (tid < 96) xreg = X[(size_t)b0 * INDIM + tid];
    __syncthreads();
    if (tid < 96) zsh[(tid / 6) * ZS + HDIM + (tid % 6)] = (_Float16)xreg;
    __syncthreads();

    // A-fragment addressing (16-bit A layout):
    //   lanes<16:  halves0..7 = K0..7  (row bytes [0:16)),  halves8..15 = K16..23 ([32:48))
    //   lanes>=16: halves0..7 = K8..15 ([16:32)),            halves8..15 = K24..31 ([48:64))
    const _Float16* zrow = zsh + (lane & 15) * ZS;
    const int aoff = (lane < 16) ? 0 : 8;      // in halves

    for (int t = 0; t < SEQ; ++t) {
        // double-buffer next timestep's x through registers
        if (t + 1 < SEQ && tid < 96)
            xreg = X[((size_t)(t + 1) * BATCH + b0) * INDIM + tid];

        AB16 a0, a1;
        a0.u[0] = *(const v4u*)(zrow +  0 + aoff);
        a0.u[1] = *(const v4u*)(zrow + 16 + aoff);
        a1.u[0] = *(const v4u*)(zrow + 32 + aoff);
        a1.u[1] = *(const v4u*)(zrow + 48 + aoff);

        v8f acc;
#pragma unroll
        for (int v = 0; v < 8; ++v) acc[v] = bias_n;

        // g = z @ W^T + bias   (K=64 as two K=32 slabs, f32 accumulate)
        acc = __builtin_amdgcn_wmma_f32_16x16x32_f16(false, a0.v, false, bfrag[0].v,
                                                     (short)0, acc, false, false);
        acc = __builtin_amdgcn_wmma_f32_16x16x32_f16(false, a1.v, false, bfrag[1].v,
                                                     (short)0, acc, false, false);

        // C layout: VGPR v -> row v (lanes<16) / v+8 (lanes>=16), col n
        {
            const int mrow = (lane < 16) ? 0 : 8;
#pragma unroll
            for (int v = 0; v < 8; ++v)
                gsh[(mrow + v) * GS + n] = acc[v];
        }
        __syncthreads();

        // gate nonlinearity + state update (c pinned in registers per thread)
        {
            const int m = tid / HDIM, j = tid % HDIM;
            const float gi = gsh[m * GS +      j];
            const float gf = gsh[m * GS + 30 + j];
            const float gg = gsh[m * GS + 60 + j];
            const float go = gsh[m * GS + 90 + j];
            c0  = fast_sigmoid(gf) * c0 + fast_sigmoid(gi) * fast_tanh(gg);
            h0f = fast_sigmoid(go) * fast_tanh(c0);
            zsh[m * ZS + j] = (_Float16)h0f;
        }
        if (e1 < 480) {
            const int m = e1 / HDIM, j = e1 % HDIM;
            const float gi = gsh[m * GS +      j];
            const float gf = gsh[m * GS + 30 + j];
            const float gg = gsh[m * GS + 60 + j];
            const float go = gsh[m * GS + 90 + j];
            c1  = fast_sigmoid(gf) * c1 + fast_sigmoid(gi) * fast_tanh(gg);
            h1f = fast_sigmoid(go) * fast_tanh(c1);
            zsh[m * ZS + j] = (_Float16)h1f;
        }
        if (t + 1 < SEQ && tid < 96)
            zsh[(tid / 6) * ZS + HDIM + (tid % 6)] = (_Float16)xreg;
        __syncthreads();
    }

    // ------------------------------------------------------------------
    // Epilogue: out = (h @ W1^T + b1) @ W2^T + b2   (tiny; plain VALU)
    // ------------------------------------------------------------------
    // stash final h as f32 in gsh cols 0..29
    {
        const int m = tid / HDIM, j = tid % HDIM;
        gsh[m * GS + j] = h0f;
    }
    if (e1 < 480) {
        const int m = e1 / HDIM, j = e1 % HDIM;
        gsh[m * GS + j] = h1f;
    }
    __syncthreads();

    // mid = h @ W1^T + b1 -> gsh cols 64..93
    {
        const int m = tid / HDIM, j = tid % HDIM;
        float s = b1[j];
#pragma unroll
        for (int k = 0; k < HDIM; ++k) s += gsh[m * GS + k] * W1[j * HDIM + k];
        gsh[m * GS + 64 + j] = s;
    }
    if (e1 < 480) {
        const int m = e1 / HDIM, j = e1 % HDIM;
        float s = b1[j];
#pragma unroll
        for (int k = 0; k < HDIM; ++k) s += gsh[m * GS + k] * W1[j * HDIM + k];
        gsh[m * GS + 64 + j] = s;
    }
    __syncthreads();

    for (int e = tid; e < 16 * OUTDIM; e += 256) {
        const int m = e / OUTDIM, o = e % OUTDIM;
        float s = b2[o];
#pragma unroll
        for (int j = 0; j < HDIM; ++j) s += gsh[m * GS + 64 + j] * W2[o * HDIM + j];
        out[(size_t)(b0 + m) * OUTDIM + o] = s;
    }
}

extern "C" void kernel_launch(void* const* d_in, const int* in_sizes, int n_in,
                              void* d_out, int out_size, void* d_ws, size_t ws_size,
                              hipStream_t stream) {
    const float* X    = (const float*)d_in[0];
    const float* W_ih = (const float*)d_in[1];
    const float* W_hh = (const float*)d_in[2];
    const float* b_ih = (const float*)d_in[3];
    const float* b_hh = (const float*)d_in[4];
    const float* W1   = (const float*)d_in[5];
    const float* b1   = (const float*)d_in[6];
    const float* W2   = (const float*)d_in[7];
    const float* b2   = (const float*)d_in[8];
    float* out = (float*)d_out;

    dim3 grid(BATCH / 16), block(256);
    hipLaunchKernelGGL(lstm_fused_kernel, grid, block, 0, stream,
                       X, W_ih, W_hh, b_ih, b_hh, W1, b1, W2, b2, out);
}
